// GlobalAttentionLayer_3410204033347
// MI455X (gfx1250) — matrix operand order repaired and checked by reading, not compiled, not run on hardware
//
#include <hip/hip_runtime.h>
#include <hip/hip_bf16.h>

typedef __attribute__((ext_vector_type(16))) _Float16 v16h;
typedef __attribute__((ext_vector_type(8)))  _Float16 v8h;
typedef __attribute__((ext_vector_type(8)))  float    v8f;

#define NN      10000          // N nodes
#define DD      256            // feature dim
#define WPR     313            // bitmask words per row (ceil(10000/32))
#define NPADK   10016          // K padded to 32*WPR
#define RPB     8              // rows per block in k0
#define BK      256            // K-chunk in main kernel
#define ASTRIDE (BK + 8)       // LDS att row stride (halfs), padded vs bank conflicts

// ---------------- Kernel 0: XW (f32 in-register), h=tanh, e = h@a, valid ----
// Also writes f16 copy of XW (rows >= NN zero-padded to NPADK).
__global__ __launch_bounds__(256) void k0_xw(const float* __restrict__ X,
                                             const float* __restrict__ W,
                                             const float* __restrict__ bias,
                                             const float* __restrict__ avec,
                                             _Float16* __restrict__ xwh,
                                             float* __restrict__ em) {
  __shared__ float xs[RPB][DD];
  __shared__ float part[RPB][8][2];
  const int t    = threadIdx.x;
  const int wave = t >> 5, lane = t & 31;
  const int r0   = blockIdx.x * RPB;

  #pragma unroll
  for (int r = 0; r < RPB; ++r) {
    int row = r0 + r;
    xs[r][t] = (row < NN) ? X[(size_t)row * DD + t] : 0.0f;
  }
  __syncthreads();

  float acc[RPB] = {};
  for (int k = 0; k < DD; ++k) {
    float w = W[(size_t)k * DD + t];
    #pragma unroll
    for (int r = 0; r < RPB; ++r) acc[r] = fmaf(xs[r][k], w, acc[r]);
  }

  const float bv = bias[t];
  const float av = avec[t];

  for (int r = 0; r < RPB; ++r) {
    int row = r0 + r;
    float xw = acc[r] + bv;
    // f16 copy for WMMA B; zero the K padding rows (10000..10015)
    if (row < NPADK) xwh[(size_t)t * NPADK + row] = (_Float16)((row < NN) ? xw : 0.0f);
    float h  = tanhf(xw);
    float hs = h, ha = h * av;
    #pragma unroll
    for (int off = 16; off > 0; off >>= 1) {
      hs += __shfl_xor(hs, off, 32);
      ha += __shfl_xor(ha, off, 32);
    }
    if (lane == 0) { part[r][wave][0] = hs; part[r][wave][1] = ha; }
  }
  __syncthreads();
  if (t < RPB) {
    int row = r0 + t;
    if (row < NN) {
      float s1 = 0.f, s2 = 0.f;
      #pragma unroll
      for (int w = 0; w < 8; ++w) { s1 += part[t][w][0]; s2 += part[t][w][1]; }
      // fold "valid" into e: invalid nodes get -3e38 -> exp underflows to exact 0
      em[row] = (s1 != 0.0f) ? s2 : -3.0e38f;
    }
  }
}

// ---------------- Kernel 1: adj -> bitmask (ballot), 400MB read once --------
__global__ __launch_bounds__(256) void k1_bits(const int* __restrict__ adj,
                                               unsigned* __restrict__ bm) {
  long long gid = (long long)blockIdx.x * 256 + threadIdx.x;
  int  lane = threadIdx.x & 31;
  long long gw = gid >> 5;                       // global word index
  if (gw >= (long long)NN * WPR) return;         // uniform per wave
  int row  = (int)(gw / WPR);
  int word = (int)(gw % WPR);
  int col  = word * 32 + lane;
  bool p   = (col < NN) && (adj[(size_t)row * NN + col] != 0);
  unsigned m = __builtin_amdgcn_ballot_w32(p);
  if (lane == 0) bm[gw] = m;
}

// ---------------- Kernel 2: per-row online softmax stats -> c = m + log(sum) -
__global__ __launch_bounds__(256) void k2_stats(const unsigned* __restrict__ bm,
                                                const float* __restrict__ em,
                                                float* __restrict__ carr) {
  const int wave = threadIdx.x >> 5, lane = threadIdx.x & 31;
  const int row  = blockIdx.x * 8 + wave;
  if (row >= NN) return;

  float m = -3.0e38f, s = 0.0f;
  const unsigned* bmrow = bm + (size_t)row * WPR;
  for (int wd = lane; wd < WPR; wd += 32) {
    unsigned bits = bmrow[wd];
    int base = wd * 32;
    while (bits) {
      int b = __builtin_ctz(bits);
      bits &= bits - 1;
      float ev = em[base + b];
      if (ev > -1.0e37f) {                    // skip invalid neighbors exactly
        if (ev > m) { s = s * __expf(m - ev) + 1.0f; m = ev; }
        else        { s += __expf(ev - m); }
      }
    }
  }
  #pragma unroll
  for (int off = 16; off > 0; off >>= 1) {
    float mo = __shfl_xor(m, off, 32);
    float so = __shfl_xor(s, off, 32);
    float M  = fmaxf(m, mo);
    s = s * __expf(m - M) + so * __expf(mo - M);   // -3e38 sentinels stay finite
    m = M;
  }
  if (lane == 0)
    carr[row] = (s > 0.0f) ? (m + __logf(s)) : 3.0e38f;  // empty row -> att == 0
}

// ---------------- Kernel 3: out = softmax(adj,e) @ XW via v_wmma ------------
// Block: 16 output rows x 256 cols; 8 waves, 2 column tiles each.
__global__ __launch_bounds__(256) void k3_main(const unsigned* __restrict__ bm,
                                               const float* __restrict__ em,
                                               const float* __restrict__ carr,
                                               const _Float16* __restrict__ xwh,
                                               float* __restrict__ out) {
  __shared__ __align__(16) _Float16 att[16 * ASTRIDE];

  const int t    = threadIdx.x;
  const int wave = t >> 5, lane = t & 31;
  const int i0   = blockIdx.x * 16;

  // builder role: row br, 16-wide K slice starting at bko
  const int br  = t & 15;
  const int bko = (t >> 4) * 16;
  const float c_r = carr[i0 + br];
  const unsigned* bmrow = bm + (size_t)(i0 + br) * WPR;

  // wmma role
  const int wr  = lane & 15;   // A row held by this lane
  const int hi  = lane >> 4;   // half-wave selector
  const int nt0 = wave * 2, nt1 = wave * 2 + 1;

  v8f acc0 = {}; v8f acc1 = {};
  union AU { v16h v; v8h h[2]; };

  for (int j0 = 0; j0 < NPADK; j0 += BK) {
    const int chunk = (NPADK - j0 < BK) ? (NPADK - j0) : BK;

    if (bko < chunk) {
      unsigned bits = bmrow[(j0 + bko) >> 5] >> (bko & 31);
      v8h lo, hh;
      #pragma unroll
      for (int q = 0; q < 8; ++q) {
        float e0 = em[j0 + bko + q];
        float e1 = em[j0 + bko + 8 + q];
        lo[q] = (_Float16)(((bits >> q)       & 1u) ? __expf(e0 - c_r) : 0.0f);
        hh[q] = (_Float16)(((bits >> (8 + q)) & 1u) ? __expf(e1 - c_r) : 0.0f);
      }
      *reinterpret_cast<v8h*>(&att[br * ASTRIDE + bko])     = lo;
      *reinterpret_cast<v8h*>(&att[br * ASTRIDE + bko + 8]) = hh;
    }
    __syncthreads();

    const int nsub = chunk >> 5;
    for (int kc = 0; kc < nsub; ++kc) {
      const int kb = kc * 32;
      // A: 16x32 f16, lane holds row wr; lanes<16: K0..7 & K16..23, lanes>=16: +8
      AU a;
      a.h[0] = *reinterpret_cast<const v8h*>(&att[wr * ASTRIDE + kb + hi * 8]);
      a.h[1] = *reinterpret_cast<const v8h*>(&att[wr * ASTRIDE + kb + 16 + hi * 8]);
      // B: 32x16 f16, lane holds col lane%16 (xwh is [DD][NPADK]); K order same as A
      const _Float16* bp = xwh + (size_t)(lane & 15) * NPADK + j0 + kb + hi * 8;
      AU u0, u1;
      u0.h[0] = *reinterpret_cast<const v8h*>(bp + (size_t)nt0 * 16 * NPADK);
      u0.h[1] = *reinterpret_cast<const v8h*>(bp + (size_t)nt0 * 16 * NPADK + 16);
      u1.h[0] = *reinterpret_cast<const v8h*>(bp + (size_t)nt1 * 16 * NPADK);
      u1.h[1] = *reinterpret_cast<const v8h*>(bp + (size_t)nt1 * 16 * NPADK + 16);
      v16h b0 = u0.v;
      v16h b1 = u1.v;
      acc0 = __builtin_amdgcn_wmma_f32_16x16x32_f16(false, a.v, false, b0,
                                                    (short)0, acc0, false, false);
      acc1 = __builtin_amdgcn_wmma_f32_16x16x32_f16(false, a.v, false, b1,
                                                    (short)0, acc1, false, false);
    }
    __syncthreads();
  }

  // C/D layout: vgpr v -> row v + 8*hi, col = lane%16
  #pragma unroll
  for (int v = 0; v < 8; ++v) {
    const size_t row = (size_t)(i0 + v + hi * 8);
    const int    col = lane & 15;
    out[row * DD + nt0 * 16 + col] = acc0[v];
    out[row * DD + nt1 * 16 + col] = acc1[v];
  }
}

// ---------------------------------------------------------------------------
extern "C" void kernel_launch(void* const* d_in, const int* in_sizes, int n_in,
                              void* d_out, int out_size, void* d_ws, size_t ws_size,
                              hipStream_t stream) {
  const float* X    = (const float*)d_in[0];
  const int*   adj  = (const int*)  d_in[1];
  // d_in[2] is M (scalar, == NN) -- shapes are compile-time constants here
  const float* W    = (const float*)d_in[3];
  const float* bias = (const float*)d_in[4];
  const float* avec = (const float*)d_in[5];
  float* out = (float*)d_out;

  char* ws = (char*)d_ws;
  _Float16* xwh = (_Float16*)ws;                         // 10016*256*2 = 5,128,192 B
  float*    em  = (float*)(ws + 5128192);                // 10016*4     =    40,064 B
  float*    cr  = (float*)(ws + 5128192 + 40064);        // 10000*4     =    40,000 B
  unsigned* bm  = (unsigned*)(ws + 5128192 + 40064 + 40000); // 10000*313*4 = 12,520,000 B

  k0_xw   <<<(NPADK + RPB - 1) / RPB, 256, 0, stream>>>(X, W, bias, avec, xwh, em);
  k1_bits <<<(int)(((long long)NN * WPR * 32 + 255) / 256), 256, 0, stream>>>(adj, bm);
  k2_stats<<<(NN + 7) / 8, 256, 0, stream>>>(bm, em, cr);
  k3_main <<<NN / 16, 256, 0, stream>>>(bm, em, cr, xwh, out);
}
